// FFJORD_72146860638599
// MI455X (gfx1250) — compile-verified, matrix operand order checked by reading, not run on hardware
//
#include <hip/hip_runtime.h>

// ---------------------------------------------------------------------------
// FFJORD forward (RK4, 8 steps) for MI455X / gfx1250, bf16 WMMA + f32 accum.
// B=8192, D=508, C=1, H=256, DIN=510 (padded to 512).
// Wave tiling: 8 waves/WG = 2 (M, 32 rows) x 4 (N, 64 cols); B fragments are
// software double-buffered so WMMAs overlap L2 weight loads.
// ---------------------------------------------------------------------------

typedef __attribute__((ext_vector_type(16))) __bf16 v16bf;
typedef __attribute__((ext_vector_type(8)))  float  v8f;
typedef __attribute__((ext_vector_type(4)))  unsigned int u32x4;

union Frag { u32x4 q[2]; v16bf v; };

__device__ __forceinline__ unsigned short f2bf(float f) {
  unsigned int u = __float_as_uint(f);
  u += 0x7FFFu + ((u >> 16) & 1u);          // round-to-nearest-even
  return (unsigned short)(u >> 16);
}
__device__ __forceinline__ float bf2f(unsigned short h) {
  return __uint_as_float(((unsigned int)h) << 16);
}

// 16x32 bf16 fragment load (A-pattern; B uses identical pattern on a
// pre-transposed [N][K] matrix). Works for LDS or global pointers.
__device__ __forceinline__ v16bf load_frag(const unsigned short* base,
                                           int row0, int stride, int k0, int lane) {
  int half = (lane >> 4) & 1, l = lane & 15;
  const unsigned short* p = base + (size_t)(row0 + l) * stride + k0 + half * 8;
  Frag f;
  f.q[0] = *(const u32x4*)(p);
  f.q[1] = *(const u32x4*)(p + 16);
  return f.v;
}

__device__ __forceinline__ v8f v8f_zero() {
  v8f z;
#pragma unroll
  for (int i = 0; i < 8; i++) z[i] = 0.0f;
  return z;
}

#define WMMA_BF16(A, B, C) \
  __builtin_amdgcn_wmma_f32_16x16x32_bf16(false, (A), false, (B), (short)0, (C), false, false)

static constexpr int BATCH = 8192;
static constexpr int DPAD  = 512;   // 508 padded

// ---------------------------------------------------------------------------
// Pipelined 32x64 tile GEMM: acc[2][4] += A[m0..m0+31][0..K) @ Bt[n0..n0+63]^T
// A row-major (LDS, bf16), Bt row-major [N][kb] (global, bf16).
// B fragments double-buffered in distinct registers so the 8 WMMAs per k-step
// overlap the next step's global loads (partial s_wait_loadcnt, no full drain).
// ---------------------------------------------------------------------------
__device__ __forceinline__ void gemm_2x4(const unsigned short* __restrict__ A,
                                         int astr,
                                         const unsigned short* __restrict__ Bt,
                                         int kb, int K, int m0, int n0, int lane,
                                         v8f acc[2][4]) {
  Frag b[4], bn[4];
#pragma unroll
  for (int j = 0; j < 4; j++) b[j].v = load_frag(Bt, n0 + 16 * j, kb, 0, lane);
  for (int k0 = 0; k0 < K; k0 += 32) {
    bool hasNext = (k0 + 32) < K;
    if (hasNext) {
#pragma unroll
      for (int j = 0; j < 4; j++)
        bn[j].v = load_frag(Bt, n0 + 16 * j, kb, k0 + 32, lane);
    }
    v16bf a0 = load_frag(A, m0, astr, k0, lane);
    v16bf a1 = load_frag(A, m0 + 16, astr, k0, lane);
#pragma unroll
    for (int j = 0; j < 4; j++) acc[0][j] = WMMA_BF16(a0, b[j].v, acc[0][j]);
#pragma unroll
    for (int j = 0; j < 4; j++) acc[1][j] = WMMA_BF16(a1, b[j].v, acc[1][j]);
    if (hasNext) {
#pragma unroll
      for (int j = 0; j < 4; j++) b[j] = bn[j];
    }
  }
}

// ---------------------------------------------------------------------------
// Weight prep: bf16 + pre-transposed layouts so B-fragments load A-style.
//  w0t[n][k] = w0[k][n]  (256x512, k>=510 -> 0)      B for xin@w0  and eps@w0
//  w1t[n][k] = w1[k][n]  (256x256)                   B for a0@w1
//  w1b[n][k] = w1[n][k]  (256x256)                   B for u1@w1^T
//  w2t[n][k] = w2[k][n]  (512x256, n>=508 -> 0)      B for a1@w2
//  w2b[n][k] = w2[n][k]  (256x512, k>=508 -> 0)      B for eps@w2^T
// ---------------------------------------------------------------------------
__global__ __launch_bounds__(256) void ffjord_prep(
    const float* __restrict__ w0, const float* __restrict__ w1,
    const float* __restrict__ w2,
    unsigned short* __restrict__ w0t, unsigned short* __restrict__ w1t,
    unsigned short* __restrict__ w1b, unsigned short* __restrict__ w2t,
    unsigned short* __restrict__ w2b) {
  int stride = gridDim.x * blockDim.x;
  int tid = blockIdx.x * blockDim.x + threadIdx.x;
  for (int i = tid; i < 256 * 512; i += stride) {
    int n = i >> 9, k = i & 511;
    w0t[i] = f2bf((k < 510) ? w0[k * 256 + n] : 0.0f);
    w2b[i] = f2bf((k < 508) ? w2[n * 508 + k] : 0.0f);
  }
  for (int i = tid; i < 256 * 256; i += stride) {
    int n = i >> 8, k = i & 255;
    w1t[i] = f2bf(w1[k * 256 + n]);
    w1b[i] = f2bf(w1[i]);
  }
  for (int i = tid; i < 512 * 256; i += stride) {
    int n = i >> 8, k = i & 255;
    w2t[i] = f2bf((n < 508) ? w2[k * 508 + n] : 0.0f);
  }
}

__global__ __launch_bounds__(256) void ffjord_init(
    const float* __restrict__ x, float* __restrict__ yPad,
    float* __restrict__ kacc, float* __restrict__ ld) {
  int stride = gridDim.x * blockDim.x;
  int tid = blockIdx.x * blockDim.x + threadIdx.x;
  for (int i = tid; i < BATCH * DPAD; i += stride) {
    int r = i >> 9, k = i & 511;
    yPad[i] = (k < 508) ? x[(size_t)r * 508 + k] : 0.0f;
    kacc[i] = 0.0f;
  }
  for (int i = tid; i < BATCH; i += stride) ld[i] = 0.0f;
}

// ---------------------------------------------------------------------------
// out[B][256] = epsPad[B][512] @ Bt^T  (Bt is [256][512] n-major, bf16).
// g2c = eps @ w2^T (Bt=w2b); v = eps @ w0[:508] (Bt=w0t; eps pad zeros kill
// w0's cond/time rows). 64 rows per WG, 8 waves (2M x 4N).
// ---------------------------------------------------------------------------
__global__ __launch_bounds__(256) void ffjord_epsgemm(
    const float* __restrict__ eps, const unsigned short* __restrict__ Bt,
    float* __restrict__ out) {
  extern __shared__ unsigned short smemE[];
  unsigned short* etile = smemE;  // [64][512] bf16
  int tid = threadIdx.x, lane = tid & 31, wave = tid >> 5;
  int rowBase = blockIdx.x * 64;
  for (int i = tid; i < 64 * 512; i += 256) {
    int r = i >> 9, k = i & 511;
    etile[i] = (k < 508) ? f2bf(eps[(size_t)(rowBase + r) * 508 + k])
                         : (unsigned short)0;
  }
  __syncthreads();
  int half = lane >> 4, l16 = lane & 15;
  int m0 = (wave & 1) * 32, n0 = (wave >> 1) * 64;
  v8f acc[2][4];
#pragma unroll
  for (int mi = 0; mi < 2; mi++)
#pragma unroll
    for (int j = 0; j < 4; j++) acc[mi][j] = v8f_zero();
  gemm_2x4(etile, 512, Bt, 512, 512, m0, n0, lane, acc);
#pragma unroll
  for (int mi = 0; mi < 2; mi++)
#pragma unroll
    for (int j = 0; j < 4; j++)
#pragma unroll
      for (int e = 0; e < 8; e++) {
        int m = m0 + mi * 16 + half * 8 + e, n = n0 + 16 * j + l16;
        out[(size_t)(rowBase + m) * 256 + n] = acc[mi][j][e];
      }
}

// ---------------------------------------------------------------------------
// Fused dyn evaluation: one RK stage.
//   xin = [y + c*kprev, cond, t, 0, 0]          (bf16, LDS)
//   a0 = tanh(xin@w0 + b0)                      (bf16, LDS)
//   a1 = tanh(a0@w1 + b1); u1 = g2c*(1-a1^2)    (bf16, LDS)
//   fy = a1@w2 + b2  -> fyout, kacc += wk*fy
//   g1 = u1@w1^T ; tr = sum_n g1*(1-a0^2)*vconst ; ld += wk*tr
// ---------------------------------------------------------------------------
__global__ __launch_bounds__(256) void ffjord_dyn(
    const float* __restrict__ yPad, const float* __restrict__ kprev,
    int useKprev, float cIn, float tVal, float wk,
    const float* __restrict__ cond,
    const unsigned short* __restrict__ w0t, const unsigned short* __restrict__ w1t,
    const unsigned short* __restrict__ w1b, const unsigned short* __restrict__ w2t,
    const float* __restrict__ b0, const float* __restrict__ b1,
    const float* __restrict__ b2,
    const float* __restrict__ g2c, const float* __restrict__ vconst,
    float* __restrict__ fyout, float* __restrict__ kacc,
    float* __restrict__ ld) {
  extern __shared__ unsigned short smem[];
  unsigned short* xin = smem;                       // [64][512] (phase 1 only)
  unsigned short* a1t = smem;                       // [64][256] (reuses xin)
  unsigned short* u1t = smem + 64 * 256;            // [64][256]
  unsigned short* a0t = smem + 64 * 512;            // [64][256]
  float* trsh = (float*)(smem + 64 * 512 + 64 * 256);  // [64]

  int tid = threadIdx.x, lane = tid & 31, wave = tid >> 5;
  int half = lane >> 4, l16 = lane & 15;
  int m0 = (wave & 1) * 32;           // 32-row slice
  int nq = wave >> 1;                 // 64-col slice index
  int rowBase = blockIdx.x * 64;

  // Phase 0: stage xin tile (y_t, cond, t, pad) as bf16.
  for (int i = tid; i < 64 * 512; i += 256) {
    int r = i >> 9, k = i & 511;
    size_t gi = (size_t)(rowBase + r) * DPAD + k;
    float val;
    if (k < 508)       val = yPad[gi] + (useKprev ? cIn * kprev[gi] : 0.0f);
    else if (k == 508) val = cond[rowBase + r];
    else if (k == 509) val = tVal;
    else               val = 0.0f;
    xin[i] = f2bf(val);
  }
  if (tid < 64) trsh[tid] = 0.0f;
  __syncthreads();

  // Phase 1: a0 = tanh(xin @ w0 + b0), K=512.
  {
    int n0 = nq * 64;
    v8f acc[2][4];
#pragma unroll
    for (int mi = 0; mi < 2; mi++)
#pragma unroll
      for (int j = 0; j < 4; j++) acc[mi][j] = v8f_zero();
    gemm_2x4(xin, 512, w0t, 512, 512, m0, n0, lane, acc);
#pragma unroll
    for (int mi = 0; mi < 2; mi++)
#pragma unroll
      for (int j = 0; j < 4; j++)
#pragma unroll
        for (int e = 0; e < 8; e++) {
          int m = m0 + mi * 16 + half * 8 + e, n = n0 + 16 * j + l16;
          float a0v = tanhf(acc[mi][j][e] + b0[n]);
          a0t[m * 256 + n] = f2bf(a0v);
        }
  }
  __syncthreads();

  // Phase 2: a1 = tanh(a0 @ w1 + b1); u1 = g2c * (1 - a1^2). K=256.
  {
    int n0 = nq * 64;
    v8f acc[2][4];
#pragma unroll
    for (int mi = 0; mi < 2; mi++)
#pragma unroll
      for (int j = 0; j < 4; j++) acc[mi][j] = v8f_zero();
    gemm_2x4(a0t, 256, w1t, 256, 256, m0, n0, lane, acc);
#pragma unroll
    for (int mi = 0; mi < 2; mi++)
#pragma unroll
      for (int j = 0; j < 4; j++)
#pragma unroll
        for (int e = 0; e < 8; e++) {
          int m = m0 + mi * 16 + half * 8 + e, n = n0 + 16 * j + l16;
          size_t row = (size_t)(rowBase + m);
          float a1v = tanhf(acc[mi][j][e] + b1[n]);
          a1t[m * 256 + n] = f2bf(a1v);
          float u1v = g2c[row * 256 + n] * (1.0f - a1v * a1v);
          u1t[m * 256 + n] = f2bf(u1v);
        }
  }
  __syncthreads();

  // Phase 3: fy = a1 @ w2 + b2 (N=508); fyout + kacc accumulation.
  for (int pass = 0; pass < 2; pass++) {
    int n0 = nq * 64 + pass * 256;
    v8f acc[2][4];
#pragma unroll
    for (int mi = 0; mi < 2; mi++)
#pragma unroll
      for (int j = 0; j < 4; j++) acc[mi][j] = v8f_zero();
    gemm_2x4(a1t, 256, w2t, 256, 256, m0, n0, lane, acc);
#pragma unroll
    for (int mi = 0; mi < 2; mi++)
#pragma unroll
      for (int j = 0; j < 4; j++)
#pragma unroll
        for (int e = 0; e < 8; e++) {
          int m = m0 + mi * 16 + half * 8 + e, n = n0 + 16 * j + l16;
          if (n < 508) {
            size_t gi = (size_t)(rowBase + m) * DPAD + n;
            float fyv = acc[mi][j][e] + b2[n];
            fyout[gi] = fyv;
            kacc[gi] += wk * fyv;
          }
        }
  }

  // Phase 4: g1 = u1 @ w1^T; trace = sum_n g1*(1-a0^2)*vconst.
  {
    int n0 = nq * 64;
    v8f acc[2][4];
#pragma unroll
    for (int mi = 0; mi < 2; mi++)
#pragma unroll
      for (int j = 0; j < 4; j++) acc[mi][j] = v8f_zero();
    gemm_2x4(u1t, 256, w1b, 256, 256, m0, n0, lane, acc);
    float pacc[2][8];
#pragma unroll
    for (int mi = 0; mi < 2; mi++)
#pragma unroll
      for (int e = 0; e < 8; e++) pacc[mi][e] = 0.0f;
#pragma unroll
    for (int mi = 0; mi < 2; mi++)
#pragma unroll
      for (int j = 0; j < 4; j++)
#pragma unroll
        for (int e = 0; e < 8; e++) {
          int m = m0 + mi * 16 + half * 8 + e, n = n0 + 16 * j + l16;
          size_t row = (size_t)(rowBase + m);
          float a0v = bf2f(a0t[m * 256 + n]);
          pacc[mi][e] += acc[mi][j][e] * (1.0f - a0v * a0v) * vconst[row * 256 + n];
        }
#pragma unroll
    for (int mi = 0; mi < 2; mi++)
#pragma unroll
      for (int e = 0; e < 8; e++)
        atomicAdd(&trsh[m0 + mi * 16 + half * 8 + e], pacc[mi][e]);
  }
  __syncthreads();
  if (tid < 64) ld[rowBase + tid] += wk * trsh[tid];
}

__global__ __launch_bounds__(256) void ffjord_update(
    float* __restrict__ yPad, float* __restrict__ kacc,
    float* __restrict__ yOut, int writeOut) {
  int stride = gridDim.x * blockDim.x;
  int tid = blockIdx.x * blockDim.x + threadIdx.x;
  for (int i = tid; i < BATCH * 508; i += stride) {
    int r = i / 508, k = i - r * 508;
    size_t gi = (size_t)r * DPAD + k;
    float y = yPad[gi] + kacc[gi];
    yPad[gi] = y;
    kacc[gi] = 0.0f;
    if (writeOut) yOut[i] = y;
  }
}

// ---------------------------------------------------------------------------
extern "C" void kernel_launch(void* const* d_in, const int* in_sizes, int n_in,
                              void* d_out, int out_size, void* d_ws, size_t ws_size,
                              hipStream_t stream) {
  const float* x    = (const float*)d_in[0];   // [8192,508]
  const float* cond = (const float*)d_in[1];   // [8192,1]
  const float* eps  = (const float*)d_in[2];   // [8192,508]
  const float* w0   = (const float*)d_in[3];   // [510,256]
  const float* b0   = (const float*)d_in[4];   // [256]
  const float* w1   = (const float*)d_in[5];   // [256,256]
  const float* b1   = (const float*)d_in[6];   // [256]
  const float* w2   = (const float*)d_in[7];   // [256,508]
  const float* b2   = (const float*)d_in[8];   // [508]
  (void)in_sizes; (void)n_in; (void)out_size; (void)ws_size;

  float* out  = (float*)d_out;
  float* yOut = out;                          // [8192*508]
  float* ld   = out + (size_t)BATCH * 508;    // [8192]

  char* ws = (char*)d_ws;
  size_t off = 0;
  auto take = [&](size_t bytes) -> char* {
    char* p = ws + off;
    off = (off + bytes + 255) & ~(size_t)255;
    return p;
  };
  unsigned short* w0t = (unsigned short*)take(256 * 512 * 2);
  unsigned short* w1t = (unsigned short*)take(256 * 256 * 2);
  unsigned short* w1b = (unsigned short*)take(256 * 256 * 2);
  unsigned short* w2t = (unsigned short*)take(512 * 256 * 2);
  unsigned short* w2b = (unsigned short*)take(256 * 512 * 2);
  float* g2c    = (float*)take((size_t)BATCH * 256 * 4);
  float* vconst = (float*)take((size_t)BATCH * 256 * 4);
  float* yPad   = (float*)take((size_t)BATCH * DPAD * 4);
  float* fyA    = (float*)take((size_t)BATCH * DPAD * 4);
  float* fyB    = (float*)take((size_t)BATCH * DPAD * 4);
  float* kacc   = (float*)take((size_t)BATCH * DPAD * 4);

  const float dt = 1.0f / 8.0f;
  const int nTiles = BATCH / 64;                       // 128 workgroups
  const size_t shE = (size_t)64 * 512 * 2;             // 64 KB
  const size_t shD = (size_t)(64 * 512 + 64 * 256) * 2 + 64 * 4;  // ~96 KB

  ffjord_prep<<<256, 256, 0, stream>>>(w0, w1, w2, w0t, w1t, w1b, w2t, w2b);
  ffjord_init<<<2048, 256, 0, stream>>>(x, yPad, kacc, ld);
  ffjord_epsgemm<<<nTiles, 256, shE, stream>>>(eps, w2b, g2c);     // eps @ w2^T
  ffjord_epsgemm<<<nTiles, 256, shE, stream>>>(eps, w0t, vconst);  // eps @ w0[:508]

  for (int s = 0; s < 8; s++) {
    float t0 = s * dt;
    // k1
    ffjord_dyn<<<nTiles, 256, shD, stream>>>(
        yPad, fyB, 0, 0.0f, t0, dt / 6.0f, cond, w0t, w1t, w1b, w2t,
        b0, b1, b2, g2c, vconst, fyA, kacc, ld);
    // k2
    ffjord_dyn<<<nTiles, 256, shD, stream>>>(
        yPad, fyA, 1, 0.5f * dt, t0 + 0.5f * dt, dt / 3.0f, cond, w0t, w1t, w1b,
        w2t, b0, b1, b2, g2c, vconst, fyB, kacc, ld);
    // k3
    ffjord_dyn<<<nTiles, 256, shD, stream>>>(
        yPad, fyB, 1, 0.5f * dt, t0 + 0.5f * dt, dt / 3.0f, cond, w0t, w1t, w1b,
        w2t, b0, b1, b2, g2c, vconst, fyA, kacc, ld);
    // k4
    ffjord_dyn<<<nTiles, 256, shD, stream>>>(
        yPad, fyA, 1, dt, t0 + dt, dt / 6.0f, cond, w0t, w1t, w1b,
        w2t, b0, b1, b2, g2c, vconst, fyB, kacc, ld);
    // y += (dt/6)(k1+2k2+2k3+k4); write output on last step
    ffjord_update<<<2048, 256, 0, stream>>>(yPad, kacc, yOut, (s == 7) ? 1 : 0);
  }
}
